// RGCNLayer_40261023432831
// MI455X (gfx1250) — matrix operand order repaired and checked by reading, not compiled, not run on hardware
//
#include <hip/hip_runtime.h>
#include <hip/hip_bf16.h>

#define NN 8192   // nodes
#define DD 128    // input dim
#define UU 128    // units
#define RR 4      // relations
#define KSPLIT 2  // k-dimension split for load balance

typedef __attribute__((ext_vector_type(16))) __bf16 v16bf;
typedef __attribute__((ext_vector_type(8)))  __bf16 v8bf;
typedef __attribute__((ext_vector_type(8)))  float  v8f;
typedef __attribute__((ext_vector_type(4)))  float  f4;

#define SLOT_STRIDE 48                    // bytes per 16-bf16 B slot (padded: bank-friendly)
#define BBUF_BYTES  (256 * SLOT_STRIDE)   // 8 tiles x 32 lanes per k-step = 12 KB

// ---------------------------------------------------------------------------
// Prep 1: Xt[d][m] = bf16(X[m][d])   (2 MB, L2 resident afterwards)
// ---------------------------------------------------------------------------
__global__ void rgcn_xt(const float* __restrict__ X, __bf16* __restrict__ Xt) {
    int gid = blockIdx.x * 256 + threadIdx.x;        // over DD*NN = 1M
    int m = gid & (NN - 1);
    int d = gid >> 13;
    Xt[gid] = (__bf16)X[(size_t)m * DD + d];
}

// ---------------------------------------------------------------------------
// Prep 2: Wt[r][u][d] = bf16(W[r][d][u])   (128 KB)
// ---------------------------------------------------------------------------
__global__ void rgcn_wt(const float* __restrict__ W, __bf16* __restrict__ Wt) {
    int gid = blockIdx.x * 256 + threadIdx.x;        // over RR*UU*DD = 64K
    int d = gid & (DD - 1);
    int u = (gid >> 7) & (UU - 1);
    int r = gid >> 14;
    Wt[gid] = (__bf16)W[((size_t)r * DD + d) * UU + u];
}

// ---------------------------------------------------------------------------
// Stage one k-step of B (Xt) into LDS, pre-swizzled into per-lane WMMA
// footprints, via async global->LDS DMA (ASYNCcnt).
// slot s = t*32 + lane  ->  u = t*16 + (s&15), k-range = ((s>>4)&1)*16
// 128 threads x 2 slots = 256 slots; each slot = 32 B via 2x b128 async.
// ---------------------------------------------------------------------------
__device__ __forceinline__ void stage_B(const __bf16* __restrict__ Xt,
                                        uint32_t ldsBase, int k0, int tid) {
    #pragma unroll
    for (int i = 0; i < 2; ++i) {
        int s  = tid * 2 + i;
        int l  = s & 31;
        int t  = s >> 5;
        int u  = t * 16 + (l & 15);
        int kb = k0 + ((l >> 4) << 4);
        uint64_t g = (uint64_t)(uintptr_t)(Xt + (size_t)u * NN + kb);
        uint32_t d = ldsBase + (uint32_t)s * SLOT_STRIDE;
        asm volatile("global_load_async_to_lds_b128 %0, %1, off"
                     :: "v"(d), "v"(g) : "memory");
        asm volatile("global_load_async_to_lds_b128 %0, %1, off offset:16"
                     :: "v"(d), "v"(g) : "memory");
    }
}

// ---------------------------------------------------------------------------
// Main: 4-wave workgroup, 128 adj rows (32 per wave), one (r, k-half) each.
//   GEMM1: msgs[32x128] += adj_r[32 x 4096] @ X[4096 x 128]   (streams adj)
//   LDS round-trip (bf16), then GEMM2: partial = msgs @ W_r
// ---------------------------------------------------------------------------
__global__ __launch_bounds__(128) void rgcn_main(
    const float*  __restrict__ adj,     // [R][N][N] fp32
    const __bf16* __restrict__ Xt,      // [D][N]
    const __bf16* __restrict__ Wt,      // [R][U][D]
    float*        __restrict__ partial) // [R*KSPLIT][N][U]
{
    __shared__ __attribute__((aligned(64))) char   Bs[2][BBUF_BYTES]; // 24 KB
    __shared__ __attribute__((aligned(64))) __bf16 Ms[4][32][DD];     // 32 KB

    const int tid  = threadIdx.x;
    const int lane = tid & 31;
    const int wv   = tid >> 5;
    const int r    = blockIdx.y;
    const int kh   = blockIdx.z;
    const int n0   = blockIdx.x * 128 + wv * 32;    // this wave's first row
    const int kbase = kh * (NN / KSPLIT);
    const int lo   = lane & 15;
    const bool hi  = lane >= 16;
    const int asub = hi ? 16 : 0;                   // A K base (contiguous, matches B)
    const int koff = hi ? 16 : 0;                   // B K base (contiguous, matches A)

    const uint32_t bsBase = (uint32_t)(uintptr_t)(&Bs[0][0]);

    v8f acc[2][8] = {};
    const float* arow0 = adj + ((size_t)r * NN + n0 + lo) * NN + kbase;
    const float* arow1 = arow0 + (size_t)16 * NN;

    // prologue: stage first B step
    stage_B(Xt, bsBase, kbase, tid);
    asm volatile("s_wait_asynccnt 0x0" ::: "memory");
    __syncthreads();

    const int KSTEPS = (NN / KSPLIT) / 32;          // 64... = 4096/32 = 128
    for (int ks = 0; ks < KSTEPS; ++ks) {
        const int buf = ks & 1;
        // prefetch next k-step into the other buffer (clamped, branchless)
        const int knext = kbase + ((ks + 1 < KSTEPS) ? (ks + 1) : ks) * 32;
        stage_B(Xt, bsBase + (uint32_t)(buf ^ 1) * BBUF_BYTES, knext, tid);

        // ---- A: 2 tiles of 16x32 fp32 -> bf16 (contiguous k, matches B) ----
        const int kk = ks * 32;
        v16bf a[2];
        #pragma unroll
        for (int at = 0; at < 2; ++at) {
            const float* ap = (at ? arow1 : arow0) + kk + asub;
            f4 a0 = *(const f4*)(ap);
            f4 a1 = *(const f4*)(ap + 4);
            f4 a2 = *(const f4*)(ap + 8);
            f4 a3 = *(const f4*)(ap + 12);
            #pragma unroll
            for (int i = 0; i < 4; ++i) {
                a[at][i]      = (__bf16)a0[i];
                a[at][4 + i]  = (__bf16)a1[i];
                a[at][8 + i]  = (__bf16)a2[i];
                a[at][12 + i] = (__bf16)a3[i];
            }
        }

        // ---- B from LDS (swizzled slots) + 16 WMMAs ----
        const char* bp = &Bs[buf][0] + (size_t)lane * SLOT_STRIDE;
        #pragma unroll
        for (int t = 0; t < 8; ++t) {
            v16bf b = *(const v16bf*)(bp + t * (32 * SLOT_STRIDE));
            acc[0][t] = __builtin_amdgcn_wmma_f32_16x16x32_bf16(
                false, a[0], false, b, (short)0, acc[0][t], false, false);
            acc[1][t] = __builtin_amdgcn_wmma_f32_16x16x32_bf16(
                false, a[1], false, b, (short)0, acc[1][t], false, false);
        }

        asm volatile("s_wait_asynccnt 0x0" ::: "memory");
        __syncthreads();
    }

    // ---------------- msgs tile -> LDS (bf16, per-wave region) ----------------
    #pragma unroll
    for (int at = 0; at < 2; ++at)
        #pragma unroll
        for (int t = 0; t < 8; ++t)
            #pragma unroll
            for (int v = 0; v < 8; ++v)
                Ms[wv][at * 16 + v + (hi ? 8 : 0)][t * 16 + lo] = (__bf16)acc[at][t][v];
    __syncthreads();

    // ---------------- GEMM2: msgs[32x128] @ W_r[128x128] ----------------
    v8f acc2[2][8] = {};
    #pragma unroll
    for (int kk2 = 0; kk2 < 4; ++kk2) {
        v16bf a2[2];
        #pragma unroll
        for (int at = 0; at < 2; ++at) {
            const __bf16* mrow = &Ms[wv][at * 16 + lo][kk2 * 32 + asub];
            v8bf m0 = *(const v8bf*)(mrow);
            v8bf m1 = *(const v8bf*)(mrow + 8);
            #pragma unroll
            for (int i = 0; i < 8; ++i) { a2[at][i] = m0[i]; a2[at][8 + i] = m1[i]; }
        }
        const int db = kk2 * 32 + koff;
        #pragma unroll
        for (int t = 0; t < 8; ++t) {
            v16bf b2 = *(const v16bf*)(Wt + ((size_t)r * UU + t * 16 + lo) * DD + db);
            acc2[0][t] = __builtin_amdgcn_wmma_f32_16x16x32_bf16(
                false, a2[0], false, b2, (short)0, acc2[0][t], false, false);
            acc2[1][t] = __builtin_amdgcn_wmma_f32_16x16x32_bf16(
                false, a2[1], false, b2, (short)0, acc2[1][t], false, false);
        }
    }

    // ---------------- store fp32 partial slice (r, k-half) ----------------
    float* pbase = partial + (size_t)(r * KSPLIT + kh) * NN * UU;
    #pragma unroll
    for (int at = 0; at < 2; ++at)
        #pragma unroll
        for (int t = 0; t < 8; ++t)
            #pragma unroll
            for (int v = 0; v < 8; ++v) {
                int row = n0 + at * 16 + v + (hi ? 8 : 0);
                pbase[(size_t)row * UU + t * 16 + lo] = acc2[at][t][v];
            }
}

// ---------------------------------------------------------------------------
// Reduce over R*KSPLIT slices + bias + ReLU
// ---------------------------------------------------------------------------
__global__ void rgcn_reduce(const float* __restrict__ partial,
                            const float* __restrict__ bias,
                            float* __restrict__ out) {
    size_t i = (size_t)blockIdx.x * 256 + threadIdx.x;   // over N*U
    int u = (int)(i & (UU - 1));
    float s = bias[u];
    #pragma unroll
    for (int p = 0; p < RR * KSPLIT; ++p) s += partial[(size_t)p * NN * UU + i];
    out[i] = fmaxf(s, 0.0f);
}

// ---------------------------------------------------------------------------
extern "C" void kernel_launch(void* const* d_in, const int* in_sizes, int n_in,
                              void* d_out, int out_size, void* d_ws, size_t ws_size,
                              hipStream_t stream) {
    const float* X    = (const float*)d_in[0];   // [N][D]
    const float* adj  = (const float*)d_in[1];   // [R][N][N]
    const float* W    = (const float*)d_in[2];   // [R][D][U]
    const float* bias = (const float*)d_in[3];   // [U]
    float* out = (float*)d_out;                  // [N][U]

    char* ws = (char*)d_ws;
    __bf16* Xt      = (__bf16*)ws;                                // 2 MiB
    __bf16* Wt      = (__bf16*)(ws + (size_t)2 * 1024 * 1024);    // 128 KiB
    float*  partial = (float*)(ws + (size_t)4 * 1024 * 1024);     // 32 MiB

    rgcn_xt<<<(DD * NN) / 256, 256, 0, stream>>>(X, Xt);
    rgcn_wt<<<(RR * UU * DD) / 256, 256, 0, stream>>>(W, Wt);
    rgcn_main<<<dim3(NN / 128, RR, KSPLIT), 128, 0, stream>>>(adj, Xt, Wt, partial);
    rgcn_reduce<<<(NN * UU) / 256, 256, 0, stream>>>(partial, bias, out);
}
